// MultiHeadAttention_23733989278287
// MI455X (gfx1250) — matrix operand order repaired and checked by reading, not compiled, not run on hardware
//
#include <hip/hip_runtime.h>
#include <hip/hip_bf16.h>

// MHA forward for MI455X (gfx1250, wave32, WMMA + TDM).
// Compute-bound (~137 GFLOP vs ~144MB HBM) -> all matmuls via
// v_wmma_f32_16x16x32_bf16. GEMM operands pre-converted to bf16 once so the
// GEMM hot loop is a pure copy pipeline: Tensor Data Mover (tensor_load_to_lds,
// TENSORcnt) double-buffers tiles into padded LDS. Softmax reductions on the
// VALU via DPP ROW_XMASK.

typedef __bf16 bf16_t;
typedef __attribute__((ext_vector_type(16))) __bf16 v16bf;
typedef __attribute__((ext_vector_type(8)))  __bf16 v8bf;
typedef __attribute__((ext_vector_type(4)))  __bf16 v4bf;
typedef __attribute__((ext_vector_type(8)))  float  v8f;

#define B_  4
#define S_  2048
#define D_  1024
#define H_  16
#define DK_ 64
#define M_  (B_ * S_)   // 8192 rows

// ---- 16-lane butterfly reductions on the VALU (DPP ROW_XMASK) -------------
template <int M>
static __device__ __forceinline__ float xm16(float x) {
  return __int_as_float(__builtin_amdgcn_update_dpp(
      __float_as_int(x), __float_as_int(x), 0x160 | M, 0xF, 0xF, false));
}
static __device__ __forceinline__ float rowmax16(float x) {
  x = fmaxf(x, xm16<1>(x));
  x = fmaxf(x, xm16<2>(x));
  x = fmaxf(x, xm16<4>(x));
  x = fmaxf(x, xm16<8>(x));
  return x;
}
static __device__ __forceinline__ float rowsum16(float x) {
  x += xm16<1>(x);
  x += xm16<2>(x);
  x += xm16<4>(x);
  x += xm16<8>(x);
  return x;
}

// Combine two 16B chunks into a v16bf fragment (2x b128 loads).
static __device__ __forceinline__ v16bf ld16(const bf16_t* p0, const bf16_t* p1) {
  v8bf lo = *(const v8bf*)p0;
  v8bf hi = *(const v8bf*)p1;
  return __builtin_shufflevector(lo, hi, 0, 1, 2, 3, 4, 5, 6, 7,
                                 8, 9, 10, 11, 12, 13, 14, 15);
}

// ---- Tensor Data Mover (CDNA5): async 2-D tile -> padded LDS ---------------
#if defined(__gfx1250__) && __has_builtin(__builtin_amdgcn_tensor_load_to_lds)
#define HAVE_TDM 1
typedef __attribute__((ext_vector_type(4))) unsigned int v4u;
typedef __attribute__((ext_vector_type(8))) int          v8i;
typedef __attribute__((ext_vector_type(4))) int          v4i;

// 2-D bf16 tile load: tile_k elems per row (must be 32 -> 16 DW rows),
// tile_rows rows, row stride in elements; pad_dw DWORDs of LDS padding
// appended after each 16-DW row (pad_interval code 3).
static __device__ __forceinline__ void tdm_load_2d(const void* gaddr,
                                                   unsigned lds_off,
                                                   int tile_k, int tile_rows,
                                                   int stride_elems,
                                                   int pad_dw) {
  unsigned long long ga = (unsigned long long)gaddr;
  v4u g0;
  g0[0] = 1u;                                    // count=1, user descriptor
  g0[1] = lds_off;                               // LDS byte address
  g0[2] = (unsigned)(ga & 0xFFFFFFFFu);          // global_addr[31:0]
  g0[3] = (unsigned)((ga >> 32) & 0x01FFFFFFu) | (2u << 30);  // [56:32]|type=2

  unsigned dw0 = (1u << 16);                     // data_size=1 (2B), mask=0
  if (pad_dw > 0) {
    dw0 |= (1u << 20);                           // pad_enable
    dw0 |= (3u << 22);                           // pad_interval: 16 DWORDs
    dw0 |= ((unsigned)(pad_dw - 1) << 25);       // pad_amount DWORDs
  }
  const unsigned td0 = (unsigned)stride_elems;   // tensor_dim0 (in-bounds)
  const unsigned td1 = 0x7FFFFFFFu;              // tensor_dim1: huge, no OOB
  v8i g1;
  g1[0] = (int)dw0;
  g1[1] = (int)((td0 & 0xFFFFu) << 16);                      // dim0 lo16
  g1[2] = (int)((td0 >> 16) | ((td1 & 0xFFFFu) << 16));      // dim0 hi|dim1 lo
  g1[3] = (int)((td1 >> 16) | ((unsigned)tile_k << 16));     // dim1 hi|tile0
  g1[4] = (int)((unsigned)tile_rows & 0xFFFFu);              // tile1, tile2=0
  g1[5] = (int)(unsigned)stride_elems;                       // dim0_stride lo
  g1[6] = 0;
  g1[7] = 0;
  v4i g2 = {1, 0, 0, 0};                         // tensor_dim2=1 (unused)
  v4i g3 = {0, 0, 0, 0};
  v8i g4 = {0, 0, 0, 0, 0, 0, 0, 0};             // extra group (6-arg form)
  __builtin_amdgcn_tensor_load_to_lds(g0, g1, g2, g3, g4, 0);
}
#else
#define HAVE_TDM 0
#endif

// ---------------------------------------------------------------------------
// Prep: f32 -> bf16 bulk convert (4 elems/thread).
// ---------------------------------------------------------------------------
__global__ __launch_bounds__(256) void cvt_bf16_kernel(
    const float* __restrict__ x, bf16_t* __restrict__ y) {
  int i = (blockIdx.x * 256 + threadIdx.x) * 4;
  float4 v = *(const float4*)&x[i];
  v4bf d;
  d[0] = (bf16_t)v.x; d[1] = (bf16_t)v.y; d[2] = (bf16_t)v.z; d[3] = (bf16_t)v.w;
  *(v4bf*)&y[i] = d;
}

// Prep: Wt[n][k] = (bf16)W[k][n]  (1024x1024, 32x32 LDS tiles).
__global__ __launch_bounds__(256) void transpose_cvt_kernel(
    const float* __restrict__ W, bf16_t* __restrict__ Wt) {
  __shared__ float T[32][33];
  const int bk = blockIdx.x * 32, bn = blockIdx.y * 32;
  const int tx = threadIdx.x & 31, ty = threadIdx.x >> 5;
  #pragma unroll
  for (int i = 0; i < 32; i += 8)
    T[ty + i][tx] = W[(size_t)(bk + ty + i) * D_ + bn + tx];
  __syncthreads();
  #pragma unroll
  for (int i = 0; i < 32; i += 8)
    Wt[(size_t)(bn + ty + i) * D_ + bk + tx] = (bf16_t)T[tx][ty + i];
}

// ---------------------------------------------------------------------------
// Tiled GEMM: out[M_,D_] = A[M_,D_] @ Wt[D_,D_]^T + bias (A, Wt bf16).
// Block = 256 threads = 8 waves; tile 256(M) x 64(N); K-step 32, double-
// buffered LDS fed by the TDM (or vectorized copies). Wave owns 32 rows:
// 2 strips x 4 N-subtiles -> 8 WMMA / K-step.
// OUT_MODE: 0 = f32 row-major [M,D]; 1 = bf16 [B,H,S,DK]; 2 = bf16 [B,H,DK,S].
// ---------------------------------------------------------------------------
template <int OUT_MODE>
__global__ __launch_bounds__(256) void mha_gemm_kernel(
    const bf16_t* __restrict__ A, const bf16_t* __restrict__ Wt,
    const float* __restrict__ bias, void* __restrict__ out_, float outScale) {
  __shared__ bf16_t As[2][256][40];  // 256x32 + 8-elem pad (rows 16 DW + 4 DW)
  __shared__ bf16_t Bt[2][64][48];   // 64x32 + 16-elem pad (rows 16 DW + 8 DW)

  const int tid  = threadIdx.x;
  const int wave = tid >> 5;
  const int lane = tid & 31;
  const int r    = lane & 15;
  const int hi   = lane >> 4;
  const int row0 = blockIdx.x * 256;
  const int n0   = blockIdx.y * 64;

  v8f acc[2][4] = {};

  // Stage one K-step (32 cols) of A (256 rows) and Wt (64 rows) into buf.
  auto stage = [&](int buf, int kk) {
#if HAVE_TDM
    if (wave == 0) {
      tdm_load_2d(&A[(size_t)row0 * D_ + kk],
                  (unsigned)(unsigned long long)&As[buf][0][0], 32, 256, D_, 4);
      tdm_load_2d(&Wt[(size_t)n0 * D_ + kk],
                  (unsigned)(unsigned long long)&Bt[buf][0][0], 32, 64, D_, 8);
    }
#else
    #pragma unroll
    for (int i = 0; i < 4; ++i) {            // A: 1024 v8bf, 4 per thread
      int f = i * 256 + tid;
      int ar = f >> 2, c8 = (f & 3) << 3;
      *(v8bf*)&As[buf][ar][c8] = *(const v8bf*)&A[(size_t)(row0 + ar) * D_ + kk + c8];
    }
    {                                        // Wt: 256 v8bf, 1 per thread
      int wn = tid >> 2, c8 = (tid & 3) << 3;
      *(v8bf*)&Bt[buf][wn][c8] = *(const v8bf*)&Wt[(size_t)(n0 + wn) * D_ + kk + c8];
    }
#endif
  };

  stage(0, 0);
  const int NK = D_ / 32;
  for (int it = 0; it < NK; ++it) {
    const int buf = it & 1;
    if (it + 1 < NK) stage(buf ^ 1, (it + 1) * 32);
#if HAVE_TDM
    if (wave == 0) {
      if (it + 1 < NK) __builtin_amdgcn_s_wait_tensorcnt(2);  // current pair done
      else             __builtin_amdgcn_s_wait_tensorcnt(0);
    }
#endif
    __syncthreads();  // buf ready for all waves

    v16bf a[2];
    #pragma unroll
    for (int s = 0; s < 2; ++s) {
      const bf16_t* ap = &As[buf][wave * 32 + s * 16 + r][0];
      a[s] = ld16(ap + hi * 16, ap + hi * 16 + 8);
    }
    #pragma unroll
    for (int j = 0; j < 4; ++j) {
      const bf16_t* bp = &Bt[buf][j * 16 + r][hi * 16];
      v16bf b = ld16(bp, bp + 8);
      #pragma unroll
      for (int s = 0; s < 2; ++s)
        acc[s][j] = __builtin_amdgcn_wmma_f32_16x16x32_bf16(
            false, a[s], false, b, (short)0, acc[s][j], false, false);
    }
    __syncthreads();  // done reading buf before it is refilled
  }

  #pragma unroll
  for (int s = 0; s < 2; ++s) {
    #pragma unroll
    for (int j = 0; j < 4; ++j) {
      #pragma unroll
      for (int e = 0; e < 8; ++e) {
        int gr = row0 + wave * 32 + s * 16 + e + 8 * hi;
        int gc = n0 + j * 16 + r;
        float v = (acc[s][j][e] + bias[gc]) * outScale;
        if (OUT_MODE == 0) {
          ((float*)out_)[(size_t)gr * D_ + gc] = v;
        } else {
          int b = gr >> 11, ss = gr & (S_ - 1);
          int h = gc >> 6,  dk = gc & (DK_ - 1);
          if (OUT_MODE == 1)
            ((bf16_t*)out_)[(((size_t)(b * H_ + h)) * S_ + ss) * DK_ + dk] = (bf16_t)v;
          else  // transposed [B,H,DK,S] for contiguous PV B-fragments
            ((bf16_t*)out_)[(((size_t)(b * H_ + h)) * DK_ + dk) * S_ + ss] = (bf16_t)v;
        }
      }
    }
  }
}

// ---------------------------------------------------------------------------
// Causal flash attention. Q/K bf16 [B,H,S,DK]; V bf16 [B,H,DK,S] (transposed).
// Wave owns 32 q-rows (2 strips); KV streamed in 32-chunks; K/V fragments
// reused across strips -> 16 WMMA per chunk. Softmax via DPP; P re-laid out
// C->A through per-wave LDS.
// ---------------------------------------------------------------------------
__global__ __launch_bounds__(256) void mha_flash_kernel(
    const bf16_t* __restrict__ Qh, const bf16_t* __restrict__ Kh,
    const bf16_t* __restrict__ Vt, bf16_t* __restrict__ Xh) {
  __shared__ bf16_t Ps[8][2][16][40];  // per-wave, per-strip 16x32 P (+pad)

  const int tid  = threadIdx.x;
  const int wave = tid >> 5;
  const int lane = tid & 31;
  const int r    = lane & 15;
  const int hi   = lane >> 4;
  const int bh   = blockIdx.y;                    // b*H + h
  const int q0   = blockIdx.x * 256 + wave * 32;  // wave's 32-row q block
  const size_t base = (size_t)bh * S_ * DK_;

  v16bf qa[2][2];
  #pragma unroll
  for (int s = 0; s < 2; ++s)
    #pragma unroll
    for (int c = 0; c < 2; ++c) {
      const bf16_t* p = Qh + base + (size_t)(q0 + s * 16 + r) * DK_ + c * 32;
      qa[s][c] = ld16(p + hi * 16, p + hi * 16 + 8);
    }

  v8f acc[2][4] = {};
  float m[2][8], l[2][8];
  #pragma unroll
  for (int s = 0; s < 2; ++s)
    #pragma unroll
    for (int e = 0; e < 8; ++e) { m[s][e] = -1e30f; l[s][e] = 0.0f; }

  const int nChunks = (q0 >> 5) + 1;  // causal: rows up to q0+31
  for (int ck = 0; ck < nChunks; ++ck) {
    const int k0 = ck * 32;

    v16bf kf[2][2];
    #pragma unroll
    for (int t = 0; t < 2; ++t)
      #pragma unroll
      for (int c = 0; c < 2; ++c) {
        const bf16_t* p =
            Kh + base + (size_t)(k0 + t * 16 + r) * DK_ + c * 32 + hi * 16;
        kf[t][c] = ld16(p, p + 8);
      }

    v8f sc[2][2] = {};
    #pragma unroll
    for (int s = 0; s < 2; ++s)
      #pragma unroll
      for (int t = 0; t < 2; ++t)
        #pragma unroll
        for (int c = 0; c < 2; ++c)
          sc[s][t] = __builtin_amdgcn_wmma_f32_16x16x32_bf16(
              false, qa[s][c], false, kf[t][c], (short)0, sc[s][t], false, false);

    #pragma unroll
    for (int s = 0; s < 2; ++s) {
      const int qs = q0 + s * 16;
      if (k0 + 31 > qs) {
        #pragma unroll
        for (int t = 0; t < 2; ++t) {
          int gc = k0 + t * 16 + r;
          #pragma unroll
          for (int e = 0; e < 8; ++e)
            if (gc > qs + e + 8 * hi) sc[s][t][e] = -1e30f;
        }
      }
      #pragma unroll
      for (int e = 0; e < 8; ++e) {
        float cm = rowmax16(fmaxf(sc[s][0][e], sc[s][1][e]));
        float mn = fmaxf(m[s][e], cm);
        float alpha = __expf(m[s][e] - mn);
        m[s][e] = mn;
        float p0 = __expf(sc[s][0][e] - mn);
        float p1 = __expf(sc[s][1][e] - mn);
        l[s][e] = l[s][e] * alpha + rowsum16(p0 + p1);
        #pragma unroll
        for (int j = 0; j < 4; ++j) acc[s][j][e] *= alpha;
        Ps[wave][s][e + 8 * hi][r]      = (bf16_t)p0;
        Ps[wave][s][e + 8 * hi][16 + r] = (bf16_t)p1;
      }
    }
    asm volatile("s_wait_dscnt 0" ::: "memory");  // DS in-order per wave

    v16bf pa[2];
    #pragma unroll
    for (int s = 0; s < 2; ++s) {
      const bf16_t* pp = &Ps[wave][s][r][0];
      pa[s] = ld16(pp + hi * 16, pp + hi * 16 + 8);
    }

    #pragma unroll
    for (int j = 0; j < 4; ++j) {
      const bf16_t* p =
          Vt + ((size_t)bh * DK_ + j * 16 + r) * S_ + k0 + hi * 16;
      v16bf vb = ld16(p, p + 8);
      #pragma unroll
      for (int s = 0; s < 2; ++s)
        acc[s][j] = __builtin_amdgcn_wmma_f32_16x16x32_bf16(
            false, pa[s], false, vb, (short)0, acc[s][j], false, false);
    }
  }

  const int b = bh >> 4, h = bh & (H_ - 1);
  #pragma unroll
  for (int s = 0; s < 2; ++s)
    #pragma unroll
    for (int e = 0; e < 8; ++e) {
      float inv = (l[s][e] > 0.0f) ? (1.0f / l[s][e]) : 0.0f;
      int gr = q0 + s * 16 + e + 8 * hi;
      #pragma unroll
      for (int j = 0; j < 4; ++j) {
        int gc = h * DK_ + j * 16 + r;
        Xh[((size_t)(b * S_ + gr)) * D_ + gc] = (bf16_t)(acc[s][j][e] * inv);
      }
    }
}

// ---------------------------------------------------------------------------
extern "C" void kernel_launch(void* const* d_in, const int* in_sizes, int n_in,
                              void* d_out, int out_size, void* d_ws,
                              size_t ws_size, hipStream_t stream) {
  const float* q  = (const float*)d_in[0];
  const float* k  = (const float*)d_in[1];
  const float* v  = (const float*)d_in[2];
  // d_in[3] = causal mask (int32) -- applied analytically in-kernel.
  const float* Wq = (const float*)d_in[4];
  const float* bq = (const float*)d_in[5];
  const float* Wk = (const float*)d_in[6];
  const float* bk = (const float*)d_in[7];
  const float* Wv = (const float*)d_in[8];
  const float* bv = (const float*)d_in[9];
  const float* Wo = (const float*)d_in[10];
  const float* bo = (const float*)d_in[11];

  const size_t tMD = (size_t)M_ * D_;      // 8M elems
  const size_t tDD = (size_t)D_ * D_;      // 1M elems
  bf16_t* qb  = (bf16_t*)d_ws;             // [M,D] bf16 inputs
  bf16_t* kb  = qb + tMD;
  bf16_t* vb  = kb + tMD;
  bf16_t* Wtq = vb + tMD;                  // [N,K] bf16 transposed weights
  bf16_t* Wtk = Wtq + tDD;
  bf16_t* Wtv = Wtk + tDD;
  bf16_t* Wto = Wtv + tDD;
  bf16_t* Qh  = Wto + tDD;                 // [B,H,S,DK]
  bf16_t* Kh  = Qh + tMD;
  bf16_t* Vt  = Kh + tMD;                  // [B,H,DK,S]
  bf16_t* Xh  = Vt + tMD;                  // [B*S, D]

  // Prep: bulk converts + weight transposes (one-time, bandwidth-trivial).
  cvt_bf16_kernel<<<tMD / 1024, 256, 0, stream>>>(q, qb);
  cvt_bf16_kernel<<<tMD / 1024, 256, 0, stream>>>(k, kb);
  cvt_bf16_kernel<<<tMD / 1024, 256, 0, stream>>>(v, vb);
  dim3 gridT(D_ / 32, D_ / 32);
  transpose_cvt_kernel<<<gridT, 256, 0, stream>>>(Wq, Wtq);
  transpose_cvt_kernel<<<gridT, 256, 0, stream>>>(Wk, Wtk);
  transpose_cvt_kernel<<<gridT, 256, 0, stream>>>(Wv, Wtv);
  transpose_cvt_kernel<<<gridT, 256, 0, stream>>>(Wo, Wto);

  dim3 gridG(M_ / 256, D_ / 64);
  const float scale = 0.125f;  // 1/sqrt(DK), folded into Q projection output
  mha_gemm_kernel<1><<<gridG, 256, 0, stream>>>(qb, Wtq, bq, Qh, scale);
  mha_gemm_kernel<1><<<gridG, 256, 0, stream>>>(kb, Wtk, bk, Kh, 1.0f);
  mha_gemm_kernel<2><<<gridG, 256, 0, stream>>>(vb, Wtv, bv, Vt, 1.0f);

  dim3 gridA(S_ / 256, B_ * H_);
  mha_flash_kernel<<<gridA, 256, 0, stream>>>(Qh, Kh, Vt, Xh);

  mha_gemm_kernel<0><<<gridG, 256, 0, stream>>>(Xh, Wto, bo, d_out, 1.0f);
}
